// TransformerLayer_18872086298869
// MI455X (gfx1250) — compile-verified, matrix operand order checked by reading, not run on hardware
//
#include <hip/hip_runtime.h>
#include <cmath>
#include <cstdint>

typedef __bf16 bf16;
typedef __attribute__((ext_vector_type(16))) __bf16 bf16x16;
typedef __attribute__((ext_vector_type(8)))  __bf16 bf16x8;
typedef __attribute__((ext_vector_type(8)))  float  f32x8;

#define LSTR 48  // LDS row stride in bf16 elements (96B: 16B-aligned b128 reads, bank spread)

// ---------------------------------------------------------------------------
// CDNA5 async global->LDS copy (ASYNCcnt-tracked), per-lane 16 bytes
// ---------------------------------------------------------------------------
__device__ __forceinline__ void async_ld_b128(unsigned lds_addr, const bf16* gsrc) {
    asm volatile("global_load_async_to_lds_b128 %0, %1, off"
                 :: "v"(lds_addr), "v"(gsrc) : "memory");
}
__device__ __forceinline__ void wait_async0() {
    asm volatile("s_wait_asynccnt 0x0" ::: "memory");
}

// ---------------------------------------------------------------------------
// LDS-tiled transpose + fp32->bf16: out[n*K + k] = (bf16)in[k*N + n]
// grid (N/32, K/32), block 256
// ---------------------------------------------------------------------------
__global__ __launch_bounds__(256)
void transpose_f32_to_bf16(const float* __restrict__ in, bf16* __restrict__ out, int K, int N) {
    __shared__ float tile[32][33];
    const int kb = blockIdx.y * 32, nb = blockIdx.x * 32;
    const int tx = threadIdx.x & 31, ty = threadIdx.x >> 5;  // ty: 0..7
#pragma unroll
    for (int r = ty; r < 32; r += 8)
        tile[r][tx] = in[(long long)(kb + r) * N + nb + tx];
    __syncthreads();
#pragma unroll
    for (int r = ty; r < 32; r += 8)
        out[(long long)(nb + r) * K + kb + tx] = (bf16)tile[tx][r];
}

// ---------------------------------------------------------------------------
// LayerNorm (unbiased std, eps added to std), fp32 in -> bf16 out
// ---------------------------------------------------------------------------
__global__ __launch_bounds__(256)
void layernorm_bf16(const float* __restrict__ x, const float* __restrict__ g,
                    const float* __restrict__ be, bf16* __restrict__ out, int D) {
    __shared__ float red[256];
    const long long row = blockIdx.x;
    const float* xr = x + row * D;
    const int t = threadIdx.x;

    float s = 0.f;
    for (int c = t; c < D; c += 256) s += xr[c];
    red[t] = s; __syncthreads();
    for (int w = 128; w > 0; w >>= 1) { if (t < w) red[t] += red[t + w]; __syncthreads(); }
    const float mean = red[0] / (float)D;
    __syncthreads();

    float s2 = 0.f;
    for (int c = t; c < D; c += 256) { float d = xr[c] - mean; s2 += d * d; }
    red[t] = s2; __syncthreads();
    for (int w = 128; w > 0; w >>= 1) { if (t < w) red[t] += red[t + w]; __syncthreads(); }
    const float var = fmaxf(red[0] / (float)(D - 1), 0.0f);
    const float denom = sqrtf(var) + 1e-6f;

    bf16* outr = out + row * D;
    for (int c = t; c < D; c += 256)
        outr[c] = (bf16)(g[c] * (xr[c] - mean) / denom + be[c]);
}

// ---------------------------------------------------------------------------
// Causal softmax, fp32 scores in -> bf16 probs out; grid (S, B)
// ---------------------------------------------------------------------------
__global__ __launch_bounds__(256)
void softmax_causal(const float* __restrict__ scores, bf16* __restrict__ P, int S) {
    __shared__ float red[256];
    const int s = blockIdx.x;
    const int b = blockIdx.y;
    const long long base = ((long long)b * S + s) * (long long)S;
    const int valid = s + 1;
    const int t = threadIdx.x;

    float mx = -3.4e38f;
    for (int c = t; c < valid; c += 256) mx = fmaxf(mx, scores[base + c]);
    red[t] = mx; __syncthreads();
    for (int w = 128; w > 0; w >>= 1) { if (t < w) red[t] = fmaxf(red[t], red[t + w]); __syncthreads(); }
    mx = red[0]; __syncthreads();

    float sum = 0.f;
    for (int c = t; c < valid; c += 256) sum += __expf(scores[base + c] - mx);
    red[t] = sum; __syncthreads();
    for (int w = 128; w > 0; w >>= 1) { if (t < w) red[t] += red[t + w]; __syncthreads(); }
    const float inv = 1.0f / red[0];

    for (int c = t; c < S; c += 256) {
        float p = (c < valid) ? __expf(scores[base + c] - mx) * inv : 0.0f;
        P[base + c] = (bf16)p;
    }
}

// ---------------------------------------------------------------------------
// NT bf16 WMMA GEMM: C = act(scale*(A @ B^T) + bias) [+ residual]
//   A: [M,K] bf16 row-major;  B: [N,K] bf16 row-major (K contiguous)
//   Block tile 128x128, K-step 32, 256 threads = 8 wave32 waves (4x2 grid),
//   each wave 32x64 = 2x4 v_wmma_f32_16x16x32_bf16.
//   Double-buffered LDS fed by global_load_async_to_lds_b128 (ASYNCcnt).
//   OUTTR: store C transposed ([N][M] per batch, contiguous along M per lane).
// ---------------------------------------------------------------------------
template<bool OUTBF16, bool RELU, bool HASBIAS, bool HASRES, bool OUTTR>
__global__ __launch_bounds__(256)
void gemm_bf16_wmma(const bf16* __restrict__ A, const bf16* __restrict__ Bm,
                    const float* __restrict__ bias, const float* __restrict__ resid,
                    void* __restrict__ Cout, int M, int N, int K,
                    long long batchA, long long batchB, long long batchC, float scale) {
    __shared__ __align__(16) bf16 sA[2][128 * LSTR];
    __shared__ __align__(16) bf16 sB[2][128 * LSTR];

    const int t    = threadIdx.x;
    const int lane = t & 31;            // wave32
    const int wave = t >> 5;            // 0..7
    const int rowW = (wave & 3) * 32;   // 4 waves along M
    const int colW = (wave >> 2) * 64;  // 2 waves along N

    const long long zb = blockIdx.z;
    A  += zb * batchA;
    Bm += zb * batchB;
    const long long cOff = zb * batchC;

    const int rowBase = blockIdx.y * 128;
    const int colBase = blockIdx.x * 128;

    f32x8 acc[2][4];
#pragma unroll
    for (int i = 0; i < 2; ++i)
#pragma unroll
        for (int j = 0; j < 4; ++j)
#pragma unroll
            for (int e = 0; e < 8; ++e) acc[i][j][e] = 0.0f;

    // ---- async staging coordinates: each thread moves 32B of A and of B ----
    const int srow = t >> 1;               // 0..127 (A: m row; B: n row)
    const int scol = (t & 1) * 16;         // 0 or 16 (k element)
    const unsigned stageOff = (unsigned)(srow * LSTR + scol) * 2u;
    const unsigned aBase = (unsigned)(uintptr_t)(&sA[0][0]) + stageOff;
    const unsigned bBase = (unsigned)(uintptr_t)(&sB[0][0]) + stageOff;
    const unsigned bufStride = 128u * LSTR * sizeof(bf16);  // 12288 bytes
    const bf16* gA = A  + (long long)(rowBase + srow) * K + scol;
    const bf16* gB = Bm + (long long)(colBase + srow) * K + scol;

    // prologue: stage first tile into buffer 0
    async_ld_b128(aBase,      gA);
    async_ld_b128(aBase + 16, gA + 8);
    async_ld_b128(bBase,      gB);
    async_ld_b128(bBase + 16, gB + 8);

    const int r16 = lane & 15;          // row/col within 16x16 tile
    const int hi8 = (lane >> 4) << 3;   // K-pair group: 0 or 8

    for (int k0 = 0; k0 < K; k0 += 32) {
        const int cur = (k0 >> 5) & 1;
        wait_async0();        // my copies into buf[cur] landed
        __syncthreads();      // everyone's copies landed; prior reads of buf[cur^1] done

        if (k0 + 32 < K) {    // stage next tile into the other buffer (overlaps WMMA)
            const unsigned nb = (unsigned)(cur ^ 1) * bufStride;
            async_ld_b128(aBase + nb,      gA + k0 + 32);
            async_ld_b128(aBase + nb + 16, gA + k0 + 40);
            async_ld_b128(bBase + nb,      gB + k0 + 32);
            async_ld_b128(bBase + nb + 16, gB + k0 + 40);
        }

        const bf16* tileA = sA[cur];
        const bf16* tileB = sB[cur];

        bf16x16 af[2];
#pragma unroll
        for (int i = 0; i < 2; ++i) {
            const bf16* p = &tileA[(rowW + i * 16 + r16) * LSTR + hi8];
            bf16x8 lo = *(const bf16x8*)(p);
            bf16x8 hi = *(const bf16x8*)(p + 16);
#pragma unroll
            for (int e = 0; e < 8; ++e) { af[i][e] = lo[e]; af[i][e + 8] = hi[e]; }
        }
        bf16x16 bfv[4];
#pragma unroll
        for (int j = 0; j < 4; ++j) {
            const bf16* p = &tileB[(colW + j * 16 + r16) * LSTR + hi8];
            bf16x8 lo = *(const bf16x8*)(p);
            bf16x8 hi = *(const bf16x8*)(p + 16);
#pragma unroll
            for (int e = 0; e < 8; ++e) { bfv[j][e] = lo[e]; bfv[j][e + 8] = hi[e]; }
        }

#pragma unroll
        for (int i = 0; i < 2; ++i)
#pragma unroll
            for (int j = 0; j < 4; ++j)
                acc[i][j] = __builtin_amdgcn_wmma_f32_16x16x32_bf16(
                    false, af[i], false, bfv[j], (short)0, acc[i][j], false, false);
    }

    // ---- epilogue: 16x16 f32 C layout: lane -> (m = r + 8*(lane>>4), n = lane&15)
    const int mb = (lane >> 4) * 8;
    const int nn = lane & 15;
#pragma unroll
    for (int i = 0; i < 2; ++i) {
#pragma unroll
        for (int j = 0; j < 4; ++j) {
            const int gn = colBase + colW + j * 16 + nn;
            const float bv = HASBIAS ? bias[gn] : 0.0f;
#pragma unroll
            for (int r = 0; r < 8; ++r) {
                const int gm = rowBase + rowW + i * 16 + mb + r;
                float v = acc[i][j][r] * scale + bv;
                if (RELU) v = fmaxf(v, 0.0f);
                const long long idx = OUTTR ? (cOff + (long long)gn * M + gm)
                                            : (cOff + (long long)gm * N + gn);
                if (HASRES) v += resid[idx];
                if (OUTBF16) ((bf16*)Cout)[idx] = (bf16)v;
                else         ((float*)Cout)[idx] = v;
            }
        }
    }
}

// ---------------------------------------------------------------------------
// Host-side orchestration
// ---------------------------------------------------------------------------
extern "C" void kernel_launch(void* const* d_in, const int* in_sizes, int n_in,
                              void* d_out, int out_size, void* d_ws, size_t ws_size,
                              hipStream_t stream) {
    (void)in_sizes; (void)n_in; (void)out_size; (void)ws_size;
    const int Bn = 4, S = 2048, D = 1024;
    const int BS = Bn * S;  // 8192 rows total

    const float* x   = (const float*)d_in[0];
    const float* wq  = (const float*)d_in[1];  const float* bq  = (const float*)d_in[2];
    const float* wk  = (const float*)d_in[3];  const float* bk  = (const float*)d_in[4];
    const float* wv  = (const float*)d_in[5];  const float* bv  = (const float*)d_in[6];
    const float* wo  = (const float*)d_in[7];  const float* bo  = (const float*)d_in[8];
    const float* w1  = (const float*)d_in[9];  const float* b1  = (const float*)d_in[10];
    const float* w2  = (const float*)d_in[11]; const float* b2  = (const float*)d_in[12];
    const float* g1  = (const float*)d_in[13]; const float* be1 = (const float*)d_in[14];
    const float* g2  = (const float*)d_in[15]; const float* be2 = (const float*)d_in[16];

    char* ws = (char*)d_ws;
    const size_t szW      = (size_t)D * D * sizeof(bf16);       //  2 MiB each
    const size_t szBSDbf  = (size_t)BS * D * sizeof(bf16);      // 16 MiB
    const size_t szScores = (size_t)Bn * S * S * sizeof(float); // 64 MiB
    const size_t szP      = (size_t)Bn * S * S * sizeof(bf16);  // 32 MiB

    // transposed bf16 weights: [out][in]
    bf16* wqb = (bf16*)(ws + 0 * szW);
    bf16* wkb = (bf16*)(ws + 1 * szW);
    bf16* wvb = (bf16*)(ws + 2 * szW);
    bf16* wob = (bf16*)(ws + 3 * szW);
    bf16* w1b = (bf16*)(ws + 4 * szW);
    bf16* w2b = (bf16*)(ws + 5 * szW);
    char* p = ws + 6 * szW;
    bf16*  xn     = (bf16*)p;  p += szBSDbf;
    bf16*  q      = (bf16*)p;  p += szBSDbf;
    bf16*  k      = (bf16*)p;  p += szBSDbf;
    bf16*  vT     = (bf16*)p;  p += szBSDbf;   // [B][D][S]
    float* scores = (float*)p; p += szScores;
    bf16*  P      = (bf16*)p;  p += szP;
    float* h      = (float*)p;
    bf16* attn = xn;  // reuse: xn dead after QKV projections
    bf16* hn   = q;   // reuse: q dead after score GEMM
    bf16* m1   = k;   // reuse: k dead after score GEMM

    const dim3 blk(256);
    const long long SD = (long long)S * D;
    const long long SS = (long long)S * S;
    const long long DS = (long long)D * S;

    // 1) weights -> bf16, transposed to [out][in] so every GEMM is NT
    {
        const dim3 g(D / 32, D / 32);
        transpose_f32_to_bf16<<<g, blk, 0, stream>>>(wq, wqb, D, D);
        transpose_f32_to_bf16<<<g, blk, 0, stream>>>(wk, wkb, D, D);
        transpose_f32_to_bf16<<<g, blk, 0, stream>>>(wv, wvb, D, D);
        transpose_f32_to_bf16<<<g, blk, 0, stream>>>(wo, wob, D, D);
        transpose_f32_to_bf16<<<g, blk, 0, stream>>>(w1, w1b, D, D);
        transpose_f32_to_bf16<<<g, blk, 0, stream>>>(w2, w2b, D, D);
    }
    // 2) LN1
    layernorm_bf16<<<dim3((unsigned)BS), blk, 0, stream>>>(x, g1, be1, xn, D);
    // 3) Q,K projections (bf16 out); V projection stores V^T per batch
    {
        const dim3 g(D / 128, BS / 128, 1);
        gemm_bf16_wmma<true, false, true, false, false><<<g, blk, 0, stream>>>(
            xn, wqb, bq, nullptr, q, BS, D, D, 0, 0, 0, 1.0f);
        gemm_bf16_wmma<true, false, true, false, false><<<g, blk, 0, stream>>>(
            xn, wkb, bk, nullptr, k, BS, D, D, 0, 0, 0, 1.0f);
    }
    {
        const dim3 g(D / 128, S / 128, Bn);
        gemm_bf16_wmma<true, false, true, false, true><<<g, blk, 0, stream>>>(
            xn, wvb, bv, nullptr, vT, S, D, D, SD, 0, DS, 1.0f);
    }
    // 4) scores = (q @ k^T) / sqrt(S)  (fp32 out, batched)
    {
        const dim3 g(S / 128, S / 128, Bn);
        gemm_bf16_wmma<false, false, false, false, false><<<g, blk, 0, stream>>>(
            q, k, nullptr, nullptr, scores, S, S, D, SD, SD, SS, 1.0f / sqrtf((float)S));
    }
    // 5) causal softmax -> P (bf16)
    softmax_causal<<<dim3(S, Bn), blk, 0, stream>>>(scores, P, S);
    // 6) attn = P @ (V^T)^T (bf16 out, batched)
    {
        const dim3 g(D / 128, S / 128, Bn);
        gemm_bf16_wmma<true, false, false, false, false><<<g, blk, 0, stream>>>(
            P, vT, nullptr, nullptr, attn, S, D, S, SS, DS, SD, 1.0f);
    }
    // 7) h = attn @ wo + bo + x (fp32, residual fused)
    {
        const dim3 g(D / 128, BS / 128, 1);
        gemm_bf16_wmma<false, false, true, true, false><<<g, blk, 0, stream>>>(
            attn, wob, bo, x, h, BS, D, D, 0, 0, 0, 1.0f);
    }
    // 8) LN2
    layernorm_bf16<<<dim3((unsigned)BS), blk, 0, stream>>>(h, g2, be2, hn, D);
    // 9) m1 = relu(hn @ w1 + b1); 10) out = m1 @ w2 + b2 + h
    {
        const dim3 g(D / 128, BS / 128, 1);
        gemm_bf16_wmma<true, true, true, false, false><<<g, blk, 0, stream>>>(
            hn, w1b, b1, nullptr, m1, BS, D, D, 0, 0, 0, 1.0f);
        gemm_bf16_wmma<false, false, true, true, false><<<g, blk, 0, stream>>>(
            m1, w2b, b2, h, (float*)d_out, BS, D, D, 0, 0, 0, 1.0f);
    }
}
